// SingleHeadAlibiAttn_25821343383726
// MI455X (gfx1250) — matrix operand order repaired and checked by reading, not compiled, not run on hardware
//
#include <hip/hip_runtime.h>
#include <hip/hip_bf16.h>
#include <math.h>

// SingleHeadAlibiAttn for MI455X (gfx1250): flash-attention with
// v_wmma_f32_16x16x32_bf16 on all three matmul stages.
//
// B=4, S=4096, D=512.  Compute-bound (~90 GFLOP vs ~100 MB HBM traffic);
// all matmuls run on the WMMA pipe in bf16 with f32 accumulation.
// Softmax finalize is done by one wave per block (not redundantly) and all
// waves prefetch the next K/V tiles (global_prefetch_b8) before the barrier ahead of it.

#define BATCH   4
#define SEQ     4096
#define D_MODEL 512
#define SLOPE   0.5f

typedef __attribute__((ext_vector_type(16))) __bf16 v16bf;
typedef __attribute__((ext_vector_type(8)))  float  v8f;

union Frag16 {            // one 16-half WMMA operand (8 VGPRs)
    v16bf v;
    uint4 q[2];           // two 16-byte chunks
};

__device__ __forceinline__ unsigned short f2bf(float f) {
    unsigned int u = __builtin_bit_cast(unsigned int, f);
    u += 0x7FFFu + ((u >> 16) & 1u);          // round-to-nearest-even
    return (unsigned short)(u >> 16);
}

// ---------------------------------------------------------------- converts
__global__ __launch_bounds__(256) void cvt_x(const float* __restrict__ x,
                                             unsigned short* __restrict__ kb,
                                             int n) {
    int i = blockIdx.x * 256 + threadIdx.x;
    if (i < n) kb[i] = f2bf(x[i]);
}

__global__ __launch_bounds__(256) void cvt_w(const float* __restrict__ wqk,
                                             const float* __restrict__ wv,
                                             unsigned short* __restrict__ wt,
                                             unsigned short* __restrict__ wvb) {
    int i = blockIdx.x * 256 + threadIdx.x;       // 0 .. 512*512-1
    int k = i / D_MODEL, n = i % D_MODEL;
    wt[n * D_MODEL + k] = f2bf(wqk[k * D_MODEL + n]);  // W_qk^T  (Bt layout)
    wvb[i]              = f2bf(wv[i]);                 // W_v as-is == Bt of W_v^T
}

// ------------------------------------------------------- Q / V projections
// One 16x16 output tile per wave; K-loop of 16 WMMAs over D=512.
__global__ __launch_bounds__(128) void prep_gemm(
    const unsigned short* __restrict__ xb,    // bf16 x   [B*S, D]
    const unsigned short* __restrict__ wt,    // bf16 Wqk^T [D, D]
    const unsigned short* __restrict__ wvb,   // bf16 Wv    [D, D]
    const float*          __restrict__ bias,  // [D]
    unsigned short*       __restrict__ qb,    // bf16 Q [B*S, D]
    unsigned short*       __restrict__ vt)    // bf16 V transposed [B][D][S]
{
    const int lane  = threadIdx.x & 31;
    const int wave  = threadIdx.x >> 5;
    const int r     = lane & 15;
    const int khalf = lane >> 4;

    const int NT = D_MODEL / 16;                  // 32 col tiles
    const int MT = (BATCH * SEQ) / 16;            // 1024 row tiles
    int tile  = blockIdx.x * 4 + wave;            // 0 .. 2*MT*NT-1
    int which = tile / (MT * NT);                 // 0 = Q, 1 = V
    int rem   = tile % (MT * NT);
    int mt    = rem / NT, nt = rem % NT;
    const int row0 = mt * 16, col0 = nt * 16;

    const unsigned short* bmat = which ? wvb : wt;
    const unsigned short* arow = xb   + (size_t)(row0 + r) * D_MODEL;
    const unsigned short* bcol = bmat + (size_t)(col0 + r) * D_MODEL;

    v8f acc = {0, 0, 0, 0, 0, 0, 0, 0};
    #pragma unroll
    for (int kk = 0; kk < D_MODEL; kk += 32) {
        Frag16 a, b;
        a.q[0] = *(const uint4*)(arow + kk + khalf * 8);        // K 0..7 / 8..15
        a.q[1] = *(const uint4*)(arow + kk + 16 + khalf * 8);   // K 16..23 / 24..31
        b.q[0] = *(const uint4*)(bcol + kk + khalf * 8);        // K 0..7 / 8..15
        b.q[1] = *(const uint4*)(bcol + kk + 16 + khalf * 8);   // K 16..23 / 24..31
        acc = __builtin_amdgcn_wmma_f32_16x16x32_bf16(
                  false, a.v, false, b.v, (short)0, acc, false, false);
    }

    if (which == 0) {
        const float qscale = 0.04419417382415922f;  // 1/sqrt(512) folded into Q
        #pragma unroll
        for (int i = 0; i < 8; ++i) {
            int row = row0 + khalf * 8 + i;
            qb[(size_t)row * D_MODEL + col0 + r] = f2bf(acc[i] * qscale);
        }
    } else {
        float bv = bias[col0 + r];
        int   bidx = row0 / SEQ;
        int   j0   = row0 % SEQ;
        #pragma unroll
        for (int i = 0; i < 8; ++i) {
            size_t e = (size_t)(col0 + r);
            vt[(size_t)bidx * D_MODEL * SEQ + e * SEQ + (j0 + khalf * 8 + i)]
                = f2bf(acc[i] + bv);
        }
    }
}

// ----------------------------------------------------------- attention core
// 1 block = 4 waves = one 16-row query tile. Wave w owns D-slice [128w,128w+128).
// Per 32-key step:
//   (1) all waves: QK^T partial WMMAs on own D-slice -> s_part; prefetch next
//       K/V tiles; barrier.
//   (2) wave 0 only: cross-wave reduce, +ALiBi, causal mask, online softmax;
//       publish bf16 P tile + per-row alpha via LDS; barrier.
//   (3) all waves: read alpha (broadcast), rescale O, PV WMMAs.
__global__ __launch_bounds__(128) void attn(
    const unsigned short* __restrict__ qb,   // bf16 Q [B*S, D]
    const unsigned short* __restrict__ kb,   // bf16 K [B*S, D]
    const unsigned short* __restrict__ vt,   // bf16 V [B][D][S]
    float*                __restrict__ out)  // f32   [B*S, D]
{
    __shared__ float s_part[4][512];                      // per-wave score partials
    __shared__ __align__(16) unsigned short s_p[16 * 32]; // P tile (bf16)
    __shared__ float s_alpha[16];                         // per-row rescale
    __shared__ float s_l[16];                             // final row sums

    const int lane  = threadIdx.x & 31;
    const int wave  = threadIdx.x >> 5;
    const int r     = lane & 15;
    const int khalf = lane >> 4;

    const int qt    = blockIdx.x;                 // 0 .. B*S/16-1
    const int b     = qt / (SEQ / 16);
    const int q0    = (qt % (SEQ / 16)) * 16;     // query base within batch
    const int grow0 = b * SEQ + q0;
    const int dbase = wave * 128;

    // Q A-fragments for this wave's D-slice, kept in registers for all key steps
    Frag16 qa[4];
    {
        const unsigned short* qrow = qb + (size_t)(grow0 + r) * D_MODEL + dbase;
        #pragma unroll
        for (int kk = 0; kk < 4; ++kk) {
            qa[kk].q[0] = *(const uint4*)(qrow + kk * 32 + khalf * 16);
            qa[kk].q[1] = *(const uint4*)(qrow + kk * 32 + khalf * 16 + 8);
        }
    }

    v8f acc[8];
    const v8f vzero = {0, 0, 0, 0, 0, 0, 0, 0};
    #pragma unroll
    for (int c = 0; c < 8; ++c) acc[c] = vzero;

    float mrow[8], lrow[8];                       // live only in wave 0
    #pragma unroll
    for (int i = 0; i < 8; ++i) { mrow[i] = -1e30f; lrow[i] = 0.0f; }

    const unsigned short* kbatch = kb + (size_t)b * SEQ * D_MODEL;
    const unsigned short* vbatch = vt + (size_t)b * D_MODEL * SEQ;
    const int jmax = ((q0 + 15) / 32) * 32;       // last 32-key step (causal)

    for (int jb = 0; jb <= jmax; jb += 32) {
        // ---- (1) QK^T partials on this wave's 128-dim slice (two 16-key tiles)
        v8f s0 = vzero, s1 = vzero;
        const unsigned short* k0 = kbatch + (size_t)(jb + r) * D_MODEL + dbase + khalf * 16;
        const unsigned short* k1 = k0 + (size_t)16 * D_MODEL;
        #pragma unroll
        for (int kk = 0; kk < 4; ++kk) {
            Frag16 b0, b1;
            b0.q[0] = *(const uint4*)(k0 + kk * 32);
            b0.q[1] = *(const uint4*)(k0 + kk * 32 + 8);
            b1.q[0] = *(const uint4*)(k1 + kk * 32);
            b1.q[1] = *(const uint4*)(k1 + kk * 32 + 8);
            s0 = __builtin_amdgcn_wmma_f32_16x16x32_bf16(
                     false, qa[kk].v, false, b0.v, (short)0, s0, false, false);
            s1 = __builtin_amdgcn_wmma_f32_16x16x32_bf16(
                     false, qa[kk].v, false, b1.v, (short)0, s1, false, false);
        }
        #pragma unroll
        for (int i = 0; i < 8; ++i) {
            s_part[wave][      i * 32 + lane] = s0[i];
            s_part[wave][256 + i * 32 + lane] = s1[i];
        }

        // prefetch next key step's K and V cachelines (hidden under barrier)
        if (jb + 32 <= jmax) {
            const unsigned short* nk = k0 + (size_t)32 * D_MODEL;
            __builtin_prefetch(nk, 0, 3);
            __builtin_prefetch(nk + (size_t)16 * D_MODEL, 0, 3);
            const unsigned short* nv =
                vbatch + (size_t)(dbase + r) * SEQ + (jb + 32) + khalf * 16;
            #pragma unroll
            for (int c = 0; c < 8; ++c)
                __builtin_prefetch(nv + (size_t)c * 16 * SEQ, 0, 3);
        }
        __syncthreads();

        // ---- (2) wave 0: finalize scores + online softmax, publish P & alpha
        if (wave == 0) {
            #pragma unroll
            for (int i = 0; i < 8; ++i) {
                int o0 = i * 32 + lane, o1 = 256 + i * 32 + lane;
                float t0 = s_part[0][o0] + s_part[1][o0] + s_part[2][o0] + s_part[3][o0];
                float t1 = s_part[0][o1] + s_part[1][o1] + s_part[2][o1] + s_part[3][o1];
                int row = q0 + khalf * 8 + i;
                int c0  = jb + r, c1 = jb + 16 + r;
                t0 += (float)(c0 - row) * SLOPE; if (c0 > row) t0 = -1e30f;  // ALiBi+causal
                t1 += (float)(c1 - row) * SLOPE; if (c1 > row) t1 = -1e30f;
                float mx = fmaxf(t0, t1);
                #pragma unroll
                for (int m = 1; m < 16; m <<= 1) mx = fmaxf(mx, __shfl_xor(mx, m, 32));
                float mnew  = fmaxf(mrow[i], mx);
                float alpha = __expf(mrow[i] - mnew);
                float p0    = __expf(t0 - mnew);
                float p1    = __expf(t1 - mnew);
                float rs = p0 + p1;
                #pragma unroll
                for (int m = 1; m < 16; m <<= 1) rs += __shfl_xor(rs, m, 32);
                lrow[i] = alpha * lrow[i] + rs;
                mrow[i] = mnew;
                int prow = khalf * 8 + i;
                s_p[prow * 32 + r]      = f2bf(p0);
                s_p[prow * 32 + 16 + r] = f2bf(p1);
                if (r == 0) s_alpha[prow] = alpha;
            }
        }
        __syncthreads();

        // ---- (3) all waves: rescale O by alpha, then O += P (16x32) @ V-slice
        #pragma unroll
        for (int i = 0; i < 8; ++i) {
            float a = s_alpha[khalf * 8 + i];     // broadcast LDS read
            #pragma unroll
            for (int c = 0; c < 8; ++c) acc[c][i] *= a;
        }
        Frag16 pa;
        pa.q[0] = *(const uint4*)(s_p + r * 32 + khalf * 16);
        pa.q[1] = *(const uint4*)(s_p + r * 32 + khalf * 16 + 8);
        const unsigned short* vp = vbatch + (size_t)(dbase + r) * SEQ + jb + khalf * 16;
        #pragma unroll
        for (int c = 0; c < 8; ++c) {
            Frag16 vb;
            vb.q[0] = *(const uint4*)(vp + (size_t)c * 16 * SEQ);
            vb.q[1] = *(const uint4*)(vp + (size_t)c * 16 * SEQ + 8);
            acc[c] = __builtin_amdgcn_wmma_f32_16x16x32_bf16(
                         false, pa.v, false, vb.v, (short)0, acc[c], false, false);
        }
    }

    // ---- publish final row sums, normalize and store f32 output
    if (wave == 0 && r == 0) {
        #pragma unroll
        for (int i = 0; i < 8; ++i) s_l[khalf * 8 + i] = lrow[i];
    }
    __syncthreads();
    #pragma unroll
    for (int i = 0; i < 8; ++i) {
        float rinv = 1.0f / s_l[khalf * 8 + i];
        size_t o = (size_t)(grow0 + khalf * 8 + i) * D_MODEL + dbase + r;
        #pragma unroll
        for (int c = 0; c < 8; ++c) out[o + c * 16] = acc[c][i] * rinv;
    }
}

// ------------------------------------------------------------------ launch
extern "C" void kernel_launch(void* const* d_in, const int* in_sizes, int n_in,
                              void* d_out, int out_size, void* d_ws, size_t ws_size,
                              hipStream_t stream) {
    const float* x   = (const float*)d_in[0];
    const float* wqk = (const float*)d_in[1];
    const float* wv  = (const float*)d_in[2];
    const float* bv  = (const float*)d_in[3];
    float* out = (float*)d_out;

    // workspace layout (bf16), ~51 MB total
    char* ws = (char*)d_ws;
    unsigned short* Qb  = (unsigned short*)(ws);                 // 16 MB
    unsigned short* Kb  = (unsigned short*)(ws + 16777216);      // 16 MB
    unsigned short* Vt  = (unsigned short*)(ws + 33554432);      // 16 MB
    unsigned short* Wt  = (unsigned short*)(ws + 50331648);      // 0.5 MB
    unsigned short* Wvb = (unsigned short*)(ws + 50855936);      // 0.5 MB

    const int nx = BATCH * SEQ * D_MODEL;                        // 8,388,608
    cvt_x<<<nx / 256, 256, 0, stream>>>(x, Kb, nx);
    cvt_w<<<(D_MODEL * D_MODEL) / 256, 256, 0, stream>>>(wqk, wv, Wt, Wvb);
    prep_gemm<<<16384, 128, 0, stream>>>(Kb, Wt, Wvb, bv, Qb, Vt); // 65536 wave-tiles
    attn<<<BATCH * SEQ / 16, 128, 0, stream>>>(Qb, Kb, Vt, out);   // 1024 blocks
}
